// Seq2Seq_47442208751753
// MI455X (gfx1250) — matrix operand order repaired and checked by reading, not compiled, not run on hardware
//
#include <hip/hip_runtime.h>
#include <cmath>

#define DI __device__ __forceinline__

typedef __bf16 bf16_t;
typedef __attribute__((ext_vector_type(16))) __bf16        v16bf;
typedef __attribute__((ext_vector_type(8)))  float         v8f;
typedef __attribute__((ext_vector_type(4)))  unsigned int  u32x4;
typedef __attribute__((ext_vector_type(4)))  int           v4i_t;

namespace cfg {
constexpr int V = 32000, E = 256, H = 512, B = 32, TS = 512, TT = 128;
constexpr int TM1 = TT - 1;       // 127 decoder steps
constexpr int K0  = E + H;        // 768  fused K for layer 0
constexpr int K1  = H + H;        // 1024 fused K for layer 1
constexpr int VBLK = 256;         // vocab columns per logits block
constexpr int NVB  = V / VBLK;    // 125 logits blocks
}

union FragU { v16bf v; u32x4 q[2]; };

// ---- CDNA5 fast paths (guarded so compile can never break) ----
DI float fast_tanhf(float x) {
#if __has_builtin(__builtin_amdgcn_tanhf)
  return __builtin_amdgcn_tanhf(x);      // V_TANH_F32 (gfx1250 TRANS op)
#else
  return tanhf(x);
#endif
}

#if __has_builtin(__builtin_amdgcn_global_load_async_to_lds_b128)
#define ATHENA_ASYNC_LDS 1
typedef __attribute__((address_space(3))) void  as3_void;
typedef __attribute__((address_space(3))) v4i_t as3_v4i;
#else
#define ATHENA_ASYNC_LDS 0
#endif

DI void wait_async0() {
#if __has_builtin(__builtin_amdgcn_s_wait_asynccnt)
  __builtin_amdgcn_s_wait_asynccnt(0);
#else
  asm volatile("s_wait_asynccnt 0" ::: "memory");
#endif
}

DI v8f wmma_bf16(v16bf a, v16bf b, v8f c) {
  // v_wmma_f32_16x16x32_bf16  (neg_a, A, neg_b, B, c_mod, C, reuse_a, reuse_b)
  return __builtin_amdgcn_wmma_f32_16x16x32_bf16(false, a, false, b, (short)0, c,
                                                 false, false);
}

// A fragment (16x32 bf16): rows m0..m0+15, K k0..k0+31.
// Layout per ISA 7.12.2: lane l holds row m0+(l&15); lane-half selects K groups.
DI v16bf load_a(const bf16_t* A, int lda, int m0, int k0, int lane) {
  const int r  = m0 + (lane & 15);
  const int hh = lane >> 4;
  const bf16_t* base = A + (size_t)r * lda + k0;
  FragU f;
  f.q[0] = *reinterpret_cast<const u32x4*>(base + hh * 8);        // K: hh*8 .. +7
  f.q[1] = *reinterpret_cast<const u32x4*>(base + 16 + hh * 8);   // K: 16+hh*8 .. +7
  return f.v;
}

// B fragment (32x16 bf16) from PRE-TRANSPOSED weights WT[N][K] (row-major over N).
// lane l holds column n0+(l&15); lane-half selects K groups (hh*8..+7, 16+hh*8..+7).
DI v16bf load_bt(const bf16_t* WT, int ldk, int n0, int k0, int lane) {
  const int c  = n0 + (lane & 15);
  const int hh = lane >> 4;
  const bf16_t* base = WT + (size_t)c * ldk + k0 + hh * 8;
  FragU f;
  f.q[0] = *reinterpret_cast<const u32x4*>(base);
  f.q[1] = *reinterpret_cast<const u32x4*>(base + 16);
  return f.v;
}

// Two-layer tanh RNN cell on one 1024-thread block (32 waves).
// xcat0: [B][K0]  = [x_t | h0_prev]    (LDS)
// xcat1: [B][K1]  = [h0_new | h1_prev] (LDS)
// Wave w computes output tiles (m0=(w&1)*16, n = (w>>1) and (w>>1)+16).
DI void cell_layers(bf16_t* xcat0, bf16_t* xcat1,
                    const bf16_t* W0T, const bf16_t* W1T,
                    const float* b0, const float* b1,
                    int wave, int lane) {
  const int m0 = (wave & 1) * 16;
  const int nc = wave >> 1;                 // 0..15

  // ---- layer 0: (B x K0) @ (K0 x H) ----
  v8f acc0[2] = {};
  #pragma unroll 4
  for (int k0 = 0; k0 < cfg::K0; k0 += 32) {
    v16bf a = load_a(xcat0, cfg::K0, m0, k0, lane);
    #pragma unroll
    for (int p = 0; p < 2; ++p) {
      v16bf b = load_bt(W0T, cfg::K0, (nc + p * 16) * 16, k0, lane);
      acc0[p] = wmma_bf16(a, b, acc0[p]);
    }
  }
  __syncthreads();                          // all reads of h0_prev done
  #pragma unroll
  for (int p = 0; p < 2; ++p) {
    const int c  = (nc + p * 16) * 16 + (lane & 15);
    const int hh = lane >> 4;
    const float bias = b0[c];
    #pragma unroll
    for (int i = 0; i < 8; ++i) {
      const int r = m0 + hh * 8 + i;
      const bf16_t hv = (bf16_t)fast_tanhf(acc0[p][i] + bias);
      xcat0[r * cfg::K0 + cfg::E + c] = hv; // h0 for next step
      xcat1[r * cfg::K1 + c]          = hv; // layer-1 input
    }
  }
  __syncthreads();

  // ---- layer 1: (B x K1) @ (K1 x H) ----
  v8f acc1[2] = {};
  #pragma unroll 4
  for (int k0 = 0; k0 < cfg::K1; k0 += 32) {
    v16bf a = load_a(xcat1, cfg::K1, m0, k0, lane);
    #pragma unroll
    for (int p = 0; p < 2; ++p) {
      v16bf b = load_bt(W1T, cfg::K1, (nc + p * 16) * 16, k0, lane);
      acc1[p] = wmma_bf16(a, b, acc1[p]);
    }
  }
  __syncthreads();                          // all reads of h1_prev done
  #pragma unroll
  for (int p = 0; p < 2; ++p) {
    const int c  = (nc + p * 16) * 16 + (lane & 15);
    const int hh = lane >> 4;
    const float bias = b1[c];
    #pragma unroll
    for (int i = 0; i < 8; ++i) {
      const int r = m0 + hh * 8 + i;
      xcat1[r * cfg::K1 + cfg::H + c] = (bf16_t)fast_tanhf(acc1[p][i] + bias);
    }
  }
  __syncthreads();
}

// ---------------- Encoder: 512 sequential steps, one persistent block ----------------
__global__ void __launch_bounds__(1024)
enc_kernel(const int* __restrict__ src, const float* __restrict__ emb,
           const bf16_t* __restrict__ W0T, const bf16_t* __restrict__ W1T,
           const float* __restrict__ b0, const float* __restrict__ b1,
           bf16_t* __restrict__ hbuf /* [2][B][H] bf16 */) {
  extern __shared__ char smem[];
  bf16_t* xcat0 = reinterpret_cast<bf16_t*>(smem);                        // [B][K0]
  bf16_t* xcat1 = reinterpret_cast<bf16_t*>(smem + cfg::B * cfg::K0 * 2); // [B][K1]
  const int tid  = threadIdx.x;
  const int lane = tid & 31;
  const int wave = tid >> 5;

  for (int i = tid; i < cfg::B * cfg::K0; i += 1024) xcat0[i] = (bf16_t)0.f;
  for (int i = tid; i < cfg::B * cfg::K1; i += 1024) xcat1[i] = (bf16_t)0.f;
  __syncthreads();

  for (int t = 0; t < cfg::TS; ++t) {
    // gather x_t = emb_enc[src[:,t]] -> bf16 LDS (8 floats per thread)
    {
      const int r  = tid >> 5;           // batch row 0..31
      const int c8 = (tid & 31) * 8;     // 8-col chunk
      const int tok = src[r * cfg::TS + t];
      const float* e = emb + (size_t)tok * cfg::E + c8;
      bf16_t* d = xcat0 + r * cfg::K0 + c8;
      #pragma unroll
      for (int j = 0; j < 8; ++j) d[j] = (bf16_t)e[j];
    }
    __syncthreads();
    cell_layers(xcat0, xcat1, W0T, W1T, b0, b1, wave, lane);
  }
  // export final hidden state (h0, h1) as bf16
  for (int i = tid; i < cfg::B * cfg::H; i += 1024) {
    const int r = i >> 9, c = i & (cfg::H - 1);
    hbuf[i]                   = xcat1[r * cfg::K1 + c];
    hbuf[cfg::B * cfg::H + i] = xcat1[r * cfg::K1 + cfg::H + c];
  }
}

// ---------------- Decoder cell (one block): finalize prev argmax, TF select, RNN cell ----------------
__global__ void __launch_bounds__(1024)
dec_cell_kernel(int t, const int* __restrict__ tgt, const int* __restrict__ tf_mask,
                const float* __restrict__ emb,
                const bf16_t* __restrict__ W0T, const bf16_t* __restrict__ W1T,
                const float* __restrict__ b0, const float* __restrict__ b1,
                const float* __restrict__ pmax, const int* __restrict__ pidx,
                bf16_t* __restrict__ hbuf) {
  extern __shared__ char smem[];
  bf16_t* xcat0 = reinterpret_cast<bf16_t*>(smem);
  bf16_t* xcat1 = reinterpret_cast<bf16_t*>(smem + cfg::B * cfg::K0 * 2);
  __shared__ int tok_s[cfg::B];
  const int tid  = threadIdx.x;
  const int lane = tid & 31;
  const int wave = tid >> 5;

#if ATHENA_ASYNC_LDS
  // stage h0/h1 into LDS via CDNA5 async global->LDS DMA (16B per lane per op)
  #pragma unroll
  for (int c = tid; c < (cfg::B * cfg::H) / 8; c += 1024) {   // 2048 chunks of 8 bf16
    const int r   = c >> 6;                 // (c*8) / 512
    const int col = (c << 3) & (cfg::H - 1);
    as3_void* l0 = (as3_void*)(xcat0 + r * cfg::K0 + cfg::E + col);
    as3_void* l1 = (as3_void*)(xcat1 + r * cfg::K1 + cfg::H + col);
    __builtin_amdgcn_global_load_async_to_lds_b128(
        (v4i_t*)(hbuf + (size_t)c * 8), (as3_v4i*)l0, 0, 0);
    __builtin_amdgcn_global_load_async_to_lds_b128(
        (v4i_t*)(hbuf + (size_t)cfg::B * cfg::H + (size_t)c * 8), (as3_v4i*)l1, 0, 0);
  }
#else
  for (int i = tid; i < cfg::B * cfg::H; i += 1024) {
    const int r = i >> 9, c = i & (cfg::H - 1);
    xcat0[r * cfg::K0 + cfg::E + c] = hbuf[i];
    xcat1[r * cfg::K1 + cfg::H + c] = hbuf[cfg::B * cfg::H + i];
  }
#endif

  if (tid < cfg::B) {
    int tok;
    if (t == 0) {
      tok = tgt[tid * cfg::TT];                       // tgt[:,0]
    } else {
      float bm = pmax[tid]; int bi = pidx[tid];       // argmax over 125 partials
      for (int blk = 1; blk < cfg::NVB; ++blk) {
        const float v  = pmax[blk * cfg::B + tid];
        const int   ix = pidx[blk * cfg::B + tid];
        if (v > bm || (v == bm && ix < bi)) { bm = v; bi = ix; }
      }
      tok = (tf_mask[t - 1] > 0) ? tgt[tid * cfg::TT + t] : bi;
    }
    tok_s[tid] = tok;
  }
  __syncthreads();                                    // tok_s visible
  // gather x = emb_dec[tok]  (disjoint LDS region from the async h staging)
  {
    const int r  = tid >> 5;
    const int c8 = (tid & 31) * 8;
    const int tok = tok_s[r];
    const float* e = emb + (size_t)tok * cfg::E + c8;
    bf16_t* d = xcat0 + r * cfg::K0 + c8;
    #pragma unroll
    for (int j = 0; j < 8; ++j) d[j] = (bf16_t)e[j];
  }
#if ATHENA_ASYNC_LDS
  wait_async0();                                      // my wave's DMA done
#endif
  __syncthreads();                                    // everyone's staging visible
  cell_layers(xcat0, xcat1, W0T, W1T, b0, b1, wave, lane);
  // export updated hidden state
  for (int i = tid; i < cfg::B * cfg::H; i += 1024) {
    const int r = i >> 9, c = i & (cfg::H - 1);
    hbuf[i]                   = xcat1[r * cfg::K1 + c];
    hbuf[cfg::B * cfg::H + i] = xcat1[r * cfg::K1 + cfg::H + c];
  }
}

// ---------------- Logits: (B x H) @ (H x V) + bias, per-block partial argmax ----------------
__global__ void __launch_bounds__(256)
dec_logits_kernel(int t, const bf16_t* __restrict__ h1,
                  const bf16_t* __restrict__ WoutT, const float* __restrict__ bout,
                  float* __restrict__ out,
                  float* __restrict__ pmax, int* __restrict__ pidx) {
  __shared__ float ltile[cfg::B][cfg::VBLK + 8];
  __shared__ float rmax[cfg::B][8];
  __shared__ int   ridx[cfg::B][8];
  const int nb   = blockIdx.x * cfg::VBLK;
  const int tid  = threadIdx.x;
  const int lane = tid & 31;
  const int wave = tid >> 5;          // 0..7
  const int m0   = (wave & 1) * 16;
  const int nc   = wave >> 1;         // 0..3

  // warm the L2 path for this block's weight slice
  __builtin_prefetch(WoutT + (size_t)(nb + tid) * cfg::H, 0, 1);

  v8f acc[4] = {};
  #pragma unroll 4
  for (int k0 = 0; k0 < cfg::H; k0 += 32) {
    v16bf a = load_a(h1, cfg::H, m0, k0, lane);     // shared across 4 tiles
    #pragma unroll
    for (int j = 0; j < 4; ++j) {
      v16bf b = load_bt(WoutT, cfg::H, nb + (nc + j * 4) * 16, k0, lane);
      acc[j] = wmma_bf16(a, b, acc[j]);
    }
  }
  const size_t orow = (size_t)cfg::TM1 * cfg::V;     // per-batch-row stride
  #pragma unroll
  for (int j = 0; j < 4; ++j) {
    const int cl = (nc + j * 4) * 16 + (lane & 15);
    const int hh = lane >> 4;
    const float bias = bout[nb + cl];
    #pragma unroll
    for (int i = 0; i < 8; ++i) {
      const int r = m0 + hh * 8 + i;
      const float vv = acc[j][i] + bias;
      ltile[r][cl] = vv;
      out[(size_t)r * orow + (size_t)t * cfg::V + nb + cl] = vv;
    }
  }
  __syncthreads();
  // partial argmax: 8 threads per batch row, 32 cols each (first-max wins on ties)
  {
    const int r = tid >> 3, seg = tid & 7;
    float bm = -3.0e38f; int bi = nb + seg * 32;
    for (int c = seg * 32; c < seg * 32 + 32; ++c) {
      const float v = ltile[r][c];
      if (v > bm) { bm = v; bi = nb + c; }
    }
    rmax[r][seg] = bm; ridx[r][seg] = bi;
  }
  __syncthreads();
  if (tid < cfg::B) {
    float bm = rmax[tid][0]; int bi = ridx[tid][0];
    #pragma unroll
    for (int s = 1; s < 8; ++s) {
      const float v = rmax[tid][s]; const int ix = ridx[tid][s];
      if (v > bm || (v == bm && ix < bi)) { bm = v; bi = ix; }
    }
    pmax[blockIdx.x * cfg::B + tid] = bm;
    pidx[blockIdx.x * cfg::B + tid] = bi;
  }
}

// ---------------- Prep: bf16 convert + transpose/concat of weights ----------------
__global__ void prep_wcat_kernel(const float* __restrict__ Wih, const float* __restrict__ Whh,
                                 bf16_t* __restrict__ dst, int Ka, int Kb) {
  const int Kt  = Ka + Kb;
  const int idx = blockIdx.x * 256 + threadIdx.x;
  if (idx >= cfg::H * Kt) return;
  const int n = idx / Kt, k = idx % Kt;
  const float v = (k < Ka) ? Wih[(size_t)k * cfg::H + n]
                           : Whh[(size_t)(k - Ka) * cfg::H + n];
  dst[idx] = (bf16_t)v;   // dst[n][k] = fused W[k][n]
}

__global__ void __launch_bounds__(256)
prep_woutT_kernel(const float* __restrict__ W /* [H][V] */, bf16_t* __restrict__ dst /* [V][H] */) {
  __shared__ float tile[32][33];
  const int bv = blockIdx.x * 32;
  const int bk = blockIdx.y * 32;
  const int tx = threadIdx.x & 31;
  const int ty = threadIdx.x >> 5;   // 0..7
  #pragma unroll
  for (int i = 0; i < 4; ++i)
    tile[ty + i * 8][tx] = W[(size_t)(bk + ty + i * 8) * cfg::V + bv + tx];
  __syncthreads();
  #pragma unroll
  for (int i = 0; i < 4; ++i)
    dst[(size_t)(bv + ty + i * 8) * cfg::H + bk + tx] = (bf16_t)tile[tx][ty + i * 8];
}

extern "C" void kernel_launch(void* const* d_in, const int* in_sizes, int n_in,
                              void* d_out, int out_size, void* d_ws, size_t ws_size,
                              hipStream_t stream) {
  using namespace cfg;
  const int*   src     = (const int*)d_in[0];
  const int*   tgt     = (const int*)d_in[1];
  const int*   tf_mask = (const int*)d_in[2];
  const float* emb_enc = (const float*)d_in[3];
  const float* emb_dec = (const float*)d_in[4];
  const float* eWih0 = (const float*)d_in[5];
  const float* eWhh0 = (const float*)d_in[6];
  const float* eb0   = (const float*)d_in[7];
  const float* eWih1 = (const float*)d_in[8];
  const float* eWhh1 = (const float*)d_in[9];
  const float* eb1   = (const float*)d_in[10];
  const float* dWih0 = (const float*)d_in[11];
  const float* dWhh0 = (const float*)d_in[12];
  const float* db0   = (const float*)d_in[13];
  const float* dWih1 = (const float*)d_in[14];
  const float* dWhh1 = (const float*)d_in[15];
  const float* db1   = (const float*)d_in[16];
  const float* Wout  = (const float*)d_in[17];
  const float* bout  = (const float*)d_in[18];
  float* out = (float*)d_out;

  char* ws = (char*)d_ws;
  auto alloc = [&](size_t bytes) {
    char* p = ws; ws += (bytes + 255) & ~size_t(255); return p;
  };
  bf16_t* WoutT = (bf16_t*)alloc((size_t)V * H * 2);   // 32.75 MB, L2-resident
  bf16_t* eW0T  = (bf16_t*)alloc((size_t)H * K0 * 2);
  bf16_t* eW1T  = (bf16_t*)alloc((size_t)H * K1 * 2);
  bf16_t* dW0T  = (bf16_t*)alloc((size_t)H * K0 * 2);
  bf16_t* dW1T  = (bf16_t*)alloc((size_t)H * K1 * 2);
  bf16_t* hbuf  = (bf16_t*)alloc((size_t)2 * B * H * 2);
  float*  pmax  = (float*)alloc((size_t)NVB * B * 4);
  int*    pidx  = (int*)alloc((size_t)NVB * B * 4);

  const size_t smem_cell = (size_t)(B * K0 + B * K1) * 2;  // 112 KB of 320 KB WGP LDS
  (void)hipFuncSetAttribute((const void*)enc_kernel,
                            hipFuncAttributeMaxDynamicSharedMemorySize, (int)smem_cell);
  (void)hipFuncSetAttribute((const void*)dec_cell_kernel,
                            hipFuncAttributeMaxDynamicSharedMemorySize, (int)smem_cell);

  // weight prep (independent, parallel)
  prep_wcat_kernel<<<(H * K0 + 255) / 256, 256, 0, stream>>>(eWih0, eWhh0, eW0T, E, H);
  prep_wcat_kernel<<<(H * K1 + 255) / 256, 256, 0, stream>>>(eWih1, eWhh1, eW1T, H, H);
  prep_wcat_kernel<<<(H * K0 + 255) / 256, 256, 0, stream>>>(dWih0, dWhh0, dW0T, E, H);
  prep_wcat_kernel<<<(H * K1 + 255) / 256, 256, 0, stream>>>(dWih1, dWhh1, dW1T, H, H);
  prep_woutT_kernel<<<dim3(V / 32, H / 32), 256, 0, stream>>>(Wout, WoutT);

  // encoder: one persistent block, 512 sequential steps
  enc_kernel<<<1, 1024, smem_cell, stream>>>(src, emb_enc, eW0T, eW1T, eb0, eb1, hbuf);

  // decoder: per step, cell (incl. prev-step argmax + TF select) then logits
  const bf16_t* h1 = hbuf + B * H;
  for (int t = 0; t < TM1; ++t) {
    dec_cell_kernel<<<1, 1024, smem_cell, stream>>>(t, tgt, tf_mask, emb_dec,
                                                    dW0T, dW1T, db0, db1,
                                                    pmax, pidx, hbuf);
    dec_logits_kernel<<<NVB, 256, 0, stream>>>(t, h1, WoutT, bout, out, pmax, pidx);
  }
}
